// DynamicRecursiveNetwork_13718125543918
// MI455X (gfx1250) — matrix operand order repaired and checked by reading, not compiled, not run on hardware
//
#include <hip/hip_runtime.h>
#include <cstdint>
#include <cstddef>

#define ND 256
#define NNODES 262144

typedef __attribute__((ext_vector_type(16))) __bf16 bf16x16;
typedef __attribute__((ext_vector_type(8)))  float  f32x8;
typedef __attribute__((ext_vector_type(16))) unsigned short u16x16;

union FragB { u16x16 u; bf16x16 bf; };
union FragA { unsigned short h[16]; bf16x16 bf; };

__device__ __forceinline__ unsigned short f32_to_bf16_rne(float f) {
  unsigned int u = __float_as_uint(f);
  u += 0x7FFFu + ((u >> 16) & 1u);
  return (unsigned short)(u >> 16);
}
__device__ __forceinline__ float bf16_to_f32(unsigned short h) {
  return __uint_as_float(((unsigned int)h) << 16);
}

// ---------------------------------------------------------------------------
// Split-precision WMMA GEMM:  out[l] = op( A[l] @ W  (+ add1[l]) (+ P[(g-1)>>3]) (+ bias) )
// A: R x 256 f32 (local rows).  W given as transposed bf16 hi/lo: WT[n*256+k].
// Workgroup = 256 threads = 8 waves covering a 64-row x 256-col tile.
// Wave w: row-tile (w&3), col-half (w>>2) -> 8 output 16x16 tiles, v8f acc each.
// Per 32-wide K step: 3 chained v_wmma_f32_16x16x32_bf16 per tile (hi*hi, hi*lo, lo*hi).
// ---------------------------------------------------------------------------
__global__ __launch_bounds__(256) void wmma_gemm256(
    const float* __restrict__ A, int R,
    const unsigned short* __restrict__ WhiT,
    const unsigned short* __restrict__ WloT,
    float* __restrict__ out,
    const float* __restrict__ add1,            // optional, indexed by local row
    const float* __restrict__ Padd,            // optional parent-gather add
    int rowStart, int parentStart,             // used with Padd
    const float* __restrict__ bias,            // optional, indexed by col
    int do_tanh)
{
  const int tid  = threadIdx.x;
  const int wave = tid >> 5;
  const int lane = tid & 31;
  const int rt   = wave & 3;
  const int cg   = wave >> 2;
  const int half = lane >> 4;
  const int l15  = lane & 15;
  const int rbase = blockIdx.x * 64 + rt * 16;

  const f32x8 vzero = {0.f,0.f,0.f,0.f,0.f,0.f,0.f,0.f};
  f32x8 acc[8];
#pragma unroll
  for (int t = 0; t < 8; ++t) acc[t] = vzero;

  const int arow = rbase + l15;               // A-fragment row for this lane
  const bool avalid = arow < R;
  const float* __restrict__ Arow = A + (size_t)arow * ND;

  for (int kk = 0; kk < 8; ++kk) {
    const int k0 = kk * 32;
    // ---- A fragment (contiguous k = 16*half + e, matching B) ----
    float av[16];
    if (avalid) {
      const float4 a0 = *(const float4*)(Arow + k0 + half * 16);
      const float4 a1 = *(const float4*)(Arow + k0 + half * 16 + 4);
      const float4 a2 = *(const float4*)(Arow + k0 + half * 16 + 8);
      const float4 a3 = *(const float4*)(Arow + k0 + half * 16 + 12);
      av[0]=a0.x; av[1]=a0.y; av[2]=a0.z;  av[3]=a0.w;
      av[4]=a1.x; av[5]=a1.y; av[6]=a1.z;  av[7]=a1.w;
      av[8]=a2.x; av[9]=a2.y; av[10]=a2.z; av[11]=a2.w;
      av[12]=a3.x; av[13]=a3.y; av[14]=a3.z; av[15]=a3.w;
    } else {
#pragma unroll
      for (int e = 0; e < 16; ++e) av[e] = 0.f;
    }
    FragA ahi, alo;
#pragma unroll
    for (int e = 0; e < 16; ++e) {
      const unsigned short hi = f32_to_bf16_rne(av[e]);
      ahi.h[e] = hi;
      alo.h[e] = f32_to_bf16_rne(av[e] - bf16_to_f32(hi));
    }
    // ---- 8 column tiles ----
#pragma unroll
    for (int ct = 0; ct < 8; ++ct) {
      const int col = cg * 128 + ct * 16 + l15;
      const size_t boff = (size_t)col * ND + (size_t)(k0 + half * 16);
      FragB bhi, blo;
      bhi.u = *(const u16x16*)(WhiT + boff);
      blo.u = *(const u16x16*)(WloT + boff);
      acc[ct] = __builtin_amdgcn_wmma_f32_16x16x32_bf16(false, ahi.bf, false, bhi.bf,
                                                        (short)0, acc[ct], false, false);
      acc[ct] = __builtin_amdgcn_wmma_f32_16x16x32_bf16(false, ahi.bf, false, blo.bf,
                                                        (short)0, acc[ct], false, false);
      acc[ct] = __builtin_amdgcn_wmma_f32_16x16x32_bf16(false, alo.bf, false, bhi.bf,
                                                        (short)0, acc[ct], false, false);
    }
  }

  // ---- epilogue (C/D layout: N = lane&15, M = r + 8*(lane>>4)) ----
#pragma unroll
  for (int ct = 0; ct < 8; ++ct) {
    const int col = cg * 128 + ct * 16 + l15;
    const float bv = bias ? bias[col] : 0.0f;
#pragma unroll
    for (int r = 0; r < 8; ++r) {
      const int l = rbase + r + 8 * half;
      if (l < R) {
        float v = acc[ct][r];
        if (add1) v += add1[(size_t)l * ND + col];
        if (Padd) {
          const int g  = rowStart + l;
          const int pi = ((g - 1) >> 3) - parentStart;
          v += Padd[(size_t)pi * ND + col];
        }
        v += bv;
        if (do_tanh) v = tanhf(v);
        out[(size_t)l * ND + col] = v;
      }
    }
  }
}

// ---------------------------------------------------------------------------
// Weight prep: split f32 W[k][n] into bf16 hi/lo, transposed to WT[n][k].
// grid = (256, 7), block = 256; blockIdx.y selects the matrix.
// ---------------------------------------------------------------------------
__global__ void prep_weights(const float* __restrict__ Wq, const float* __restrict__ Wk,
                             const float* __restrict__ Wv, const float* __restrict__ Wx,
                             const float* __restrict__ Wc, const float* __restrict__ Wr,
                             const float* __restrict__ Wp,
                             unsigned short* __restrict__ WhiT,
                             unsigned short* __restrict__ WloT)
{
  const int idx = blockIdx.x * blockDim.x + threadIdx.x;   // 0..65535 == k*256+n
  const int m = blockIdx.y;
  const float* W;
  switch (m) {
    case 0: W = Wq; break; case 1: W = Wk; break; case 2: W = Wv; break;
    case 3: W = Wx; break; case 4: W = Wc; break; case 5: W = Wr; break;
    default: W = Wp; break;
  }
  const int k = idx >> 8, n = idx & 255;
  const float f = W[idx];
  const unsigned short hi = f32_to_bf16_rne(f);
  const unsigned short lo = f32_to_bf16_rne(f - bf16_to_f32(hi));
  const size_t o = (size_t)m * 65536 + (size_t)n * ND + k;
  WhiT[o] = hi;
  WloT[o] = lo;
}

// Leaves: h = tanh(xb) over rows [leaf0, N), vectorized float4
__global__ void leaves_tanh(const float* __restrict__ xb, float* __restrict__ h,
                            int leaf0, int n4)
{
  const int i = blockIdx.x * 256 + threadIdx.x;
  if (i >= n4) return;
  const size_t base = (size_t)leaf0 * ND;
  float4 v = ((const float4*)(xb + base))[i];
  v.x = tanhf(v.x); v.y = tanhf(v.y); v.z = tanhf(v.z); v.w = tanhf(v.w);
  ((float4*)(h + base))[i] = v;
}

// ---------------------------------------------------------------------------
// Attention: one wave32 per parent p; children are contiguous 8p+1..8p+8.
// Each lane owns 8 contiguous d's; dot reduced across the wave via shfl_xor.
// ctx[p-P0] = softmax_c( q[p]·k[c] / sqrt(D) ) · v[c]   (0 if no children)
// ---------------------------------------------------------------------------
__global__ __launch_bounds__(256) void attn_kernel(
    const float* __restrict__ q, const float* __restrict__ kbuf,
    const float* __restrict__ vbuf, float* __restrict__ ctx,
    int P0, int P1, int C0)
{
  const int wave = threadIdx.x >> 5;
  const int lane = threadIdx.x & 31;
  const int p = P0 + blockIdx.x * 8 + wave;
  if (p >= P1) return;
  const int d0 = lane * 8;

  const float4 qa = *(const float4*)(q + (size_t)p * ND + d0);
  const float4 qb = *(const float4*)(q + (size_t)p * ND + d0 + 4);

  float logit[8];
  bool  valid[8];
#pragma unroll
  for (int c = 0; c < 8; ++c) {
    const int node = 8 * p + 1 + c;
    valid[c] = node < NNODES;
    float part = 0.f;
    if (valid[c]) {
      const float* kr = kbuf + (size_t)(node - C0) * ND + d0;
      const float4 ka = *(const float4*)kr;
      const float4 kb = *(const float4*)(kr + 4);
      part = qa.x*ka.x + qa.y*ka.y + qa.z*ka.z + qa.w*ka.w
           + qb.x*kb.x + qb.y*kb.y + qb.z*kb.z + qb.w*kb.w;
    }
#pragma unroll
    for (int off = 16; off > 0; off >>= 1)
      part += __shfl_xor(part, off, 32);
    logit[c] = part * 0.0625f;                 // 1/sqrt(256)
  }

  float m = -3.0e38f;
#pragma unroll
  for (int c = 0; c < 8; ++c) if (valid[c]) m = fmaxf(m, logit[c]);

  float w[8], denom = 0.f;
#pragma unroll
  for (int c = 0; c < 8; ++c) {
    w[c] = valid[c] ? expf(logit[c] - m) : 0.f;
    denom += w[c];
  }
  const float inv = denom > 0.f ? 1.0f / fmaxf(denom, 1e-9f) : 0.f;

  float c8[8] = {0.f,0.f,0.f,0.f,0.f,0.f,0.f,0.f};
#pragma unroll
  for (int c = 0; c < 8; ++c) {
    if (valid[c]) {
      const int node = 8 * p + 1 + c;
      const float* vr = vbuf + (size_t)(node - C0) * ND + d0;
      const float4 va = *(const float4*)vr;
      const float4 vb = *(const float4*)(vr + 4);
      c8[0] += w[c]*va.x; c8[1] += w[c]*va.y; c8[2] += w[c]*va.z; c8[3] += w[c]*va.w;
      c8[4] += w[c]*vb.x; c8[5] += w[c]*vb.y; c8[6] += w[c]*vb.z; c8[7] += w[c]*vb.w;
    }
  }
  float* orow = ctx + (size_t)(p - P0) * ND + d0;
  float4 o0 = { c8[0]*inv, c8[1]*inv, c8[2]*inv, c8[3]*inv };
  float4 o1 = { c8[4]*inv, c8[5]*inv, c8[6]*inv, c8[7]*inv };
  *(float4*)orow       = o0;
  *(float4*)(orow + 4) = o1;
}

// ---------------------------------------------------------------------------
extern "C" void kernel_launch(void* const* d_in, const int* in_sizes, int n_in,
                              void* d_out, int out_size, void* d_ws, size_t ws_size,
                              hipStream_t stream)
{
  (void)in_sizes; (void)n_in; (void)out_size; (void)ws_size;
  const float* x  = (const float*)d_in[0];
  const float* Wq = (const float*)d_in[1];
  const float* Wk = (const float*)d_in[2];
  const float* Wv = (const float*)d_in[3];
  const float* Wx = (const float*)d_in[4];
  const float* Wc = (const float*)d_in[5];
  const float* b  = (const float*)d_in[6];
  const float* Wr = (const float*)d_in[7];
  const float* Wp = (const float*)d_in[8];
  const float* br = (const float*)d_in[9];
  float* hfin = (float*)d_out;

  // Level start indices: S(l) = (8^l - 1)/7, last clamped to N. depth(i)=l for i in [S[l],S[l+1])
  static const int S[8] = {0, 1, 9, 73, 585, 4681, 37449, 262144};

  char* ws = (char*)d_ws;
  size_t off = 0;
  auto alloc = [&](size_t bytes) -> char* {
    char* p = ws + off;
    off += (bytes + 255) & ~(size_t)255;
    return p;
  };
  unsigned short* WhiT = (unsigned short*)alloc((size_t)7 * 65536 * 2);
  unsigned short* WloT = (unsigned short*)alloc((size_t)7 * 65536 * 2);
  float* q   = (float*)alloc((size_t)37504  * ND * 4);   // q for internal rows
  float* xb  = (float*)alloc((size_t)NNODES * ND * 4);   // x@Wx + b
  float* hbu = (float*)alloc((size_t)NNODES * ND * 4);   // bottom-up h
  float* kb  = (float*)alloc((size_t)224768 * ND * 4);   // k of child rows
  float* vb  = (float*)alloc((size_t)224768 * ND * 4);   // v of child rows
  float* cp  = (float*)alloc((size_t)32768  * ND * 4);   // ctx / parent-msg buffer

#define W_HI(m) (WhiT + (size_t)(m) * 65536)
#define W_LO(m) (WloT + (size_t)(m) * 65536)
  // matrix planes: 0=Wq 1=Wk 2=Wv 3=Wx 4=Wc 5=Wr 6=Wp
  prep_weights<<<dim3(256, 7), 256, 0, stream>>>(Wq, Wk, Wv, Wx, Wc, Wr, Wp, WhiT, WloT);

  // xb = x@Wx + b   (all rows)
  wmma_gemm256<<<(NNODES + 63) / 64, 256, 0, stream>>>(
      x, NNODES, W_HI(3), W_LO(3), xb, nullptr, nullptr, 0, 0, b, 0);
  // q = x@Wq        (internal rows only)
  wmma_gemm256<<<(S[6] + 63) / 64, 256, 0, stream>>>(
      x, S[6], W_HI(0), W_LO(0), q, nullptr, nullptr, 0, 0, nullptr, 0);
  // leaves: h = tanh(xb)
  {
    const int n4 = (NNODES - S[6]) * (ND / 4);
    leaves_tanh<<<(n4 + 255) / 256, 256, 0, stream>>>(xb, hbu, S[6], n4);
  }

  // ---- bottom-up ----
  for (int lvl = 5; lvl >= 0; --lvl) {
    const int C0 = S[lvl + 1], C1 = S[lvl + 2], CR = C1 - C0;   // child rows (depth lvl+1)
    const int P0 = S[lvl],     P1 = S[lvl + 1], PR = P1 - P0;   // parent rows (depth lvl)
    wmma_gemm256<<<(CR + 63) / 64, 256, 0, stream>>>(
        hbu + (size_t)C0 * ND, CR, W_HI(1), W_LO(1), kb, nullptr, nullptr, 0, 0, nullptr, 0);
    wmma_gemm256<<<(CR + 63) / 64, 256, 0, stream>>>(
        hbu + (size_t)C0 * ND, CR, W_HI(2), W_LO(2), vb, nullptr, nullptr, 0, 0, nullptr, 0);
    attn_kernel<<<(PR + 7) / 8, 256, 0, stream>>>(q, kb, vb, cp, P0, P1, C0);
    // h[parent] = tanh(ctx@Wc + xb[parent])
    wmma_gemm256<<<(PR + 63) / 64, 256, 0, stream>>>(
        cp, PR, W_HI(4), W_LO(4), hbu + (size_t)P0 * ND,
        xb + (size_t)P0 * ND, nullptr, 0, 0, nullptr, 1);
  }

  // ---- top-down ----
  // root: h_out[0] = tanh(h[0]@Wr + br)
  wmma_gemm256<<<1, 256, 0, stream>>>(
      hbu, 1, W_HI(5), W_LO(5), hfin, nullptr, nullptr, 0, 0, br, 1);
  for (int lvl = 1; lvl <= 6; ++lvl) {
    const int Q0 = S[lvl - 1], Q1 = S[lvl], QR = Q1 - Q0;       // parent rows (updated)
    const int C0 = S[lvl], C1 = S[lvl + 1], CR = C1 - C0;       // this level's rows
    // P = h_out[parents] @ Wp
    wmma_gemm256<<<(QR + 63) / 64, 256, 0, stream>>>(
        hfin + (size_t)Q0 * ND, QR, W_HI(6), W_LO(6), cp, nullptr, nullptr, 0, 0, nullptr, 0);
    // h_out[c] = tanh(h_bu[c]@Wr + P[(c-1)>>3] + br)
    wmma_gemm256<<<(CR + 63) / 64, 256, 0, stream>>>(
        hbu + (size_t)C0 * ND, CR, W_HI(5), W_LO(5), hfin + (size_t)C0 * ND,
        nullptr, cp, C0, Q0, br, 1);
  }
#undef W_HI
#undef W_LO
}
